// Quantizer1D_12618613915789
// MI455X (gfx1250) — matrix operand order repaired and checked by reading, not compiled, not run on hardware
//
#include <hip/hip_runtime.h>

// ---------------------------------------------------------------------------
// VQ quantizer for MI455X (gfx1250).
// Exploits: encoder output depends only on token value t in [0,512) ->
// precompute 512 z-vectors, WMMA-based 512x512 nearest-code search,
// then the big [B,T] pass is a pure bandwidth-bound gather (~68MB out).
// ---------------------------------------------------------------------------

typedef __attribute__((ext_vector_type(16))) _Float16 v16h;
typedef __attribute__((ext_vector_type(8)))  _Float16 h8;
typedef __attribute__((ext_vector_type(8)))  float    v8f;

#define KCODES 512
#define DDIM   64
#define HDIM   64
#define NTOK   (16 * 16384)   // B*T = 262144

// workspace byte offsets (total ~394 KB)
#define WS_ZH   0u            // 512*64 f16  (64 KB)  z hi
#define WS_ZL   65536u        // 512*64 f16  (64 KB)  z lo residual
#define WS_CBH  131072u       // 512*64 f16  (64 KB)  codebook hi
#define WS_CBL  196608u       // 512*64 f16  (64 KB)  codebook lo residual
#define WS_ZN   262144u       // 512 f32     (2 KB)   ||z||^2
#define WS_CN   264192u       // 512 f32     (2 KB)   ||e||^2
#define WS_MAP  266240u       // 512 i32     (2 KB)   nearest code per token value
#define WS_ERR  268288u       // 512 f32     (2 KB)   ||q - z||^2 per token value
#define WS_QT   270336u       // 512*64 f32  (128 KB) quantized vector table
#define WS_HIST 401408u       // 512 u32     (2 KB)   token-value histogram

// ---------------------------------------------------------------------------
// Kernel 1: encoder for all 512 token values. One thread per token value.
// Also zeroes the histogram for this call (deterministic re-launch).
// ---------------------------------------------------------------------------
__global__ void vq_encoder_kernel(const float* __restrict__ W1, const float* __restrict__ b1,
                                  const float* __restrict__ g,  const float* __restrict__ bb,
                                  const float* __restrict__ W2, const float* __restrict__ b2,
                                  _Float16* __restrict__ zh, _Float16* __restrict__ zl,
                                  float* __restrict__ znorm, unsigned* __restrict__ hist) {
    const int v = blockIdx.x * blockDim.x + threadIdx.x;
    if (v >= KCODES) return;

    const float norm = (float)v * (2.0f / (float)(KCODES - 1)) - 1.0f;

    float hr[HDIM];
    float s = 0.f;
    #pragma unroll
    for (int j = 0; j < HDIM; ++j) { hr[j] = norm * W1[j] + b1[j]; s += hr[j]; }
    const float mu = s * (1.0f / (float)HDIM);
    float s2 = 0.f;
    #pragma unroll
    for (int j = 0; j < HDIM; ++j) { const float d = hr[j] - mu; s2 += d * d; }
    const float rstd = rsqrtf(s2 * (1.0f / (float)HDIM) + 1e-5f);

    float acc[DDIM];
    #pragma unroll
    for (int d = 0; d < DDIM; ++d) acc[d] = b2[d];
    for (int j = 0; j < HDIM; ++j) {
        float hn = (hr[j] - mu) * rstd * g[j] + bb[j];
        hn = fmaxf(hn, 0.0f);
        const float* __restrict__ w2r = W2 + j * DDIM;
        #pragma unroll
        for (int d = 0; d < DDIM; ++d) acc[d] += hn * w2r[d];
    }

    float zn = 0.f;
    #pragma unroll
    for (int d = 0; d < DDIM; ++d) {
        const float z = acc[d];
        zn += z * z;
        const _Float16 ah = (_Float16)z;
        zh[v * DDIM + d] = ah;
        zl[v * DDIM + d] = (_Float16)(z - (float)ah);   // hi/lo split
    }
    znorm[v] = zn;
    hist[v] = 0u;
}

// ---------------------------------------------------------------------------
// Kernel 2: codebook -> f16 hi/lo split + ||e||^2. One thread per code.
// ---------------------------------------------------------------------------
__global__ void vq_codebook_kernel(const float* __restrict__ cb,
                                   _Float16* __restrict__ cbh, _Float16* __restrict__ cbl,
                                   float* __restrict__ cnorm) {
    const int v = blockIdx.x * blockDim.x + threadIdx.x;
    if (v >= KCODES) return;
    float s = 0.f;
    #pragma unroll
    for (int d = 0; d < DDIM; ++d) {
        const float c = cb[v * DDIM + d];
        s += c * c;
        const _Float16 ch = (_Float16)c;
        cbh[v * DDIM + d] = ch;
        cbl[v * DDIM + d] = (_Float16)(c - (float)ch);
    }
    cnorm[v] = s;
}

// ---------------------------------------------------------------------------
// Kernel 3: WMMA nearest-neighbor search. One wave (32 threads) per 16-token
// M-tile; loops over 32 code N-tiles. dot(z,e) via v_wmma_f32_16x16x32_f16
// with hi/lo splits: zh*ch + zl*ch + zh*cl per K-step (K=64 -> 2 steps).
// argmin over  cnorm[k] - 2*dot  (||z||^2 is row-constant).
// ---------------------------------------------------------------------------
__global__ void vq_search_kernel(const _Float16* __restrict__ zh, const _Float16* __restrict__ zl,
                                 const _Float16* __restrict__ cbh, const _Float16* __restrict__ cbl,
                                 const float* __restrict__ znorm, const float* __restrict__ cnorm,
                                 int* __restrict__ map, float* __restrict__ err) {
    const int lane = threadIdx.x;   // 0..31, wave32
    const int mm = lane & 15;
    const int hp = lane >> 4;       // half-wave: 0 or 1
    const int m0 = blockIdx.x * 16; // token tile base

    // --- A fragments (16x32 f16 layout): lane holds row mm;
    //     halves 0..15 -> K = kb + 16*hp + i (contiguous, matches B)
    const _Float16* __restrict__ zhrow = zh + (m0 + mm) * DDIM;
    const _Float16* __restrict__ zlrow = zl + (m0 + mm) * DDIM;
    v16h azh0, azh1, azl0, azl1;
    {
        h8 p, q;
        p = *(const h8*)(zhrow + 16 * hp);      q = *(const h8*)(zhrow + 8 + 16 * hp);
        #pragma unroll
        for (int i = 0; i < 8; ++i) { azh0[i] = p[i]; azh0[i + 8] = q[i]; }
        p = *(const h8*)(zhrow + 32 + 16 * hp); q = *(const h8*)(zhrow + 40 + 16 * hp);
        #pragma unroll
        for (int i = 0; i < 8; ++i) { azh1[i] = p[i]; azh1[i + 8] = q[i]; }
        p = *(const h8*)(zlrow + 16 * hp);      q = *(const h8*)(zlrow + 8 + 16 * hp);
        #pragma unroll
        for (int i = 0; i < 8; ++i) { azl0[i] = p[i]; azl0[i + 8] = q[i]; }
        p = *(const h8*)(zlrow + 32 + 16 * hp); q = *(const h8*)(zlrow + 40 + 16 * hp);
        #pragma unroll
        for (int i = 0; i < 8; ++i) { azl1[i] = p[i]; azl1[i + 8] = q[i]; }
    }

    float best[8];
    int   bidx[8];
    #pragma unroll
    for (int r = 0; r < 8; ++r) { best[r] = 3.4e38f; bidx[r] = 0x7fffffff; }

    for (int nt = 0; nt < 32; ++nt) {
        const int nbase = nt * 16;
        // --- B fragments (32x16): lane holds column mm; K = kb + 16*hp + i (16
        //     contiguous halves = one aligned 32B load)
        const _Float16* __restrict__ chrow = cbh + (nbase + mm) * DDIM;
        const _Float16* __restrict__ clrow = cbl + (nbase + mm) * DDIM;
        const v16h bh0 = *(const v16h*)(chrow + 16 * hp);
        const v16h bh1 = *(const v16h*)(chrow + 32 + 16 * hp);
        const v16h bl0 = *(const v16h*)(clrow + 16 * hp);
        const v16h bl1 = *(const v16h*)(clrow + 32 + 16 * hp);

        v8f acc = {0.f, 0.f, 0.f, 0.f, 0.f, 0.f, 0.f, 0.f};
        acc = __builtin_amdgcn_wmma_f32_16x16x32_f16(false, azh0, false, bh0, (short)0, acc, false, false);
        acc = __builtin_amdgcn_wmma_f32_16x16x32_f16(false, azl0, false, bh0, (short)0, acc, false, false);
        acc = __builtin_amdgcn_wmma_f32_16x16x32_f16(false, azh0, false, bl0, (short)0, acc, false, false);
        acc = __builtin_amdgcn_wmma_f32_16x16x32_f16(false, azh1, false, bh1, (short)0, acc, false, false);
        acc = __builtin_amdgcn_wmma_f32_16x16x32_f16(false, azl1, false, bh1, (short)0, acc, false, false);
        acc = __builtin_amdgcn_wmma_f32_16x16x32_f16(false, azh1, false, bl1, (short)0, acc, false, false);

        // C layout: element r of this lane is (M = r + 8*hp, N = mm)
        const float cn = cnorm[nbase + mm];
        const int code = nbase + mm;
        #pragma unroll
        for (int r = 0; r < 8; ++r) {
            const float d = cn - 2.0f * acc[r];
            if (d < best[r] || (d == best[r] && code < bidx[r])) { best[r] = d; bidx[r] = code; }
        }
    }

    // reduce the 16 lanes sharing each M (xor masks 1,2,4,8 stay within halves)
    #pragma unroll
    for (int r = 0; r < 8; ++r) {
        float b = best[r];
        int   i = bidx[r];
        for (int m = 1; m < 16; m <<= 1) {
            const float ob = __shfl_xor(b, m, 32);
            const int   oi = __shfl_xor(i, m, 32);
            if (ob < b || (ob == b && oi < i)) { b = ob; i = oi; }
        }
        best[r] = b; bidx[r] = i;
    }

    if (mm == 0) {
        #pragma unroll
        for (int r = 0; r < 8; ++r) {
            const int m = m0 + hp * 8 + r;
            map[m] = bidx[r];
            err[m] = znorm[m] + best[r];  // ||z||^2 + (||e||^2 - 2 z.e) = ||q-z||^2
        }
    }
}

// ---------------------------------------------------------------------------
// Kernel 4: q_table[v] = codebook[map[v]]  (512 x 64 f32, float4 copies)
// ---------------------------------------------------------------------------
__global__ void vq_qtable_kernel(const float4* __restrict__ cb4, const int* __restrict__ map,
                                 float4* __restrict__ qt4) {
    const int e = blockIdx.x * blockDim.x + threadIdx.x;  // 0..8191
    if (e >= KCODES * 16) return;
    const int v = e >> 4, d4 = e & 15;
    qt4[e] = cb4[map[v] * 16 + d4];
}

// ---------------------------------------------------------------------------
// Kernel 5: the big bandwidth pass. 16 tokens per 256-thread block; each group
// of 16 lanes streams one token's 64 floats as coalesced float4s. Lane d4==0
// writes idx (as float) and bumps the integer histogram (deterministic).
// ---------------------------------------------------------------------------
__global__ void vq_gather_kernel(const int* __restrict__ t, const float4* __restrict__ qt4,
                                 const int* __restrict__ map, float4* __restrict__ out_q,
                                 float* __restrict__ out_idx, unsigned* __restrict__ hist) {
    const int tid = threadIdx.x;
    const int n = blockIdx.x * 16 + (tid >> 4);
    const int d4 = tid & 15;
    const int tv = t[n];
    out_q[n * 16 + d4] = qt4[tv * 16 + d4];
    if (d4 == 0) {
        out_idx[n] = (float)map[tv];
        atomicAdd(&hist[tv], 1u);
    }
}

// ---------------------------------------------------------------------------
// Kernel 6: loss = 1.25/(N*D) * sum_v hist[v]*err[v], fixed summation order.
// ---------------------------------------------------------------------------
__global__ void vq_loss_kernel(const unsigned* __restrict__ hist, const float* __restrict__ err,
                               float* __restrict__ loss_out) {
    __shared__ float sh[KCODES];
    const int v = threadIdx.x;
    sh[v] = (float)hist[v] * err[v];
    __syncthreads();
    if (v == 0) {
        float s = 0.f;
        for (int i = 0; i < KCODES; ++i) s += sh[i];
        loss_out[0] = s * (1.25f / ((float)NTOK * (float)DDIM));
    }
}

// ---------------------------------------------------------------------------
extern "C" void kernel_launch(void* const* d_in, const int* in_sizes, int n_in,
                              void* d_out, int out_size, void* d_ws, size_t ws_size,
                              hipStream_t stream) {
    const int*   t   = (const int*)d_in[0];     // [B,T,1] int32
    const float* W1  = (const float*)d_in[1];   // [1,H]
    const float* b1  = (const float*)d_in[2];   // [H]
    const float* g   = (const float*)d_in[3];   // ln_g [H]
    const float* bb  = (const float*)d_in[4];   // ln_b [H]
    const float* W2  = (const float*)d_in[5];   // [H,D]
    const float* b2  = (const float*)d_in[6];   // [D]
    const float* cb  = (const float*)d_in[7];   // [K,D]

    char* ws = (char*)d_ws;
    _Float16* zh    = (_Float16*)(ws + WS_ZH);
    _Float16* zl    = (_Float16*)(ws + WS_ZL);
    _Float16* cbh   = (_Float16*)(ws + WS_CBH);
    _Float16* cbl   = (_Float16*)(ws + WS_CBL);
    float*    znorm = (float*)(ws + WS_ZN);
    float*    cnorm = (float*)(ws + WS_CN);
    int*      map   = (int*)(ws + WS_MAP);
    float*    err   = (float*)(ws + WS_ERR);
    float4*   qt4   = (float4*)(ws + WS_QT);
    unsigned* hist  = (unsigned*)(ws + WS_HIST);

    float* out      = (float*)d_out;
    float4* out_q   = (float4*)out;                     // [N,64] f32
    float* out_idx  = out + (size_t)NTOK * DDIM;        // [N] (idx as float)
    float* out_loss = out_idx + NTOK;                   // scalar

    vq_encoder_kernel<<<2, 256, 0, stream>>>(W1, b1, g, bb, W2, b2, zh, zl, znorm, hist);
    vq_codebook_kernel<<<2, 256, 0, stream>>>(cb, cbh, cbl, cnorm);
    vq_search_kernel<<<32, 32, 0, stream>>>(zh, zl, cbh, cbl, znorm, cnorm, map, err);
    vq_qtable_kernel<<<32, 256, 0, stream>>>((const float4*)cb, map, qt4);
    vq_gather_kernel<<<NTOK / 16, 256, 0, stream>>>(t, qt4, map, out_q, out_idx, hist);
    vq_loss_kernel<<<1, KCODES, 0, stream>>>(hist, err, out_loss);
}
